// CWICDense_62388694941874
// MI455X (gfx1250) — matrix operand order repaired and checked by reading, not compiled, not run on hardware
//
#include <hip/hip_runtime.h>

// ---------------------------------------------------------------------------
// CWICDense eval forward on MI455X (gfx1250):
//   y[m, n*128+q] = sum_k gate_n(x_off[m,k]) * W[k, n*128+q] + out_mu
// v3: bf16 hi/lo split-GEMM (3x v_wmma_f32_16x16x32_bf16) with
//   - pre-pass kernels: X -> (x-mu) split hi/lo bf16 (interleaved pairs);
//                       W -> transposed hi/lo bf16 planes (contiguous)
//   - B tiles staged to LDS by the Tensor Data Mover (tensor_load_to_lds +
//     s_wait_tensorcnt) when the 5-arg builtin is available (clang-22/ROCm),
//     else via global_load_async_to_lds_b128 (ASYNCcnt)
//   - double-buffered LDS, one barrier per K-tile
//   - fallback (self-contained) kernel if d_ws is too small
// ---------------------------------------------------------------------------

typedef __bf16        v16bf __attribute__((ext_vector_type(16)));
typedef float         v8f   __attribute__((ext_vector_type(8)));
typedef unsigned int  v8u   __attribute__((ext_vector_type(8)));
typedef unsigned int  v4u   __attribute__((ext_vector_type(4)));
typedef int           v8i   __attribute__((ext_vector_type(8)));
typedef int           v4i   __attribute__((ext_vector_type(4)));

#define DIM_I   4096   // in_features (= K)
#define DIM_O   4096   // out_features (= N_total)
#define DIM_M   1024   // B*S rows
#define NSTRIPE 32
#define QW      128    // stripe width

#define MT 128         // block tile M
#define NT 128         // block tile N (one stripe)
#define KT 64          // K per LDS stage
#define KU (KT / 2)    // packed uints (bf16 pairs) per row
#define LDP (KU + 1)   // padded LDS row stride (uints) -> TDM pad 1dw / 32dw

// TDM builtin: 5-arg on ROCm 7.2 (clang-22); clang-23 has a 6-arg variant with
// different semantics -> use the async-load path there instead.
#if defined(__has_builtin)
#  if __has_builtin(__builtin_amdgcn_tensor_load_to_lds) && (__clang_major__ < 23)
#    define CWIC_USE_TDM 1
#  endif
#endif
#ifndef CWIC_USE_TDM
#  define CWIC_USE_TDM 0
#endif

__device__ __forceinline__ unsigned short f2bf(float f) {
    unsigned int u = __builtin_bit_cast(unsigned int, f);
    u += 0x7fffu + ((u >> 16) & 1u);          // round-to-nearest-even
    return (unsigned short)(u >> 16);
}
__device__ __forceinline__ float bf2f(unsigned short h) {
    unsigned int u = ((unsigned int)h) << 16;
    return __builtin_bit_cast(float, u);
}
__device__ __forceinline__ unsigned int pack2(unsigned short a, unsigned short b) {
    return (unsigned int)a | ((unsigned int)b << 16);
}

// A-matrix 16x32 bf16 fragment: lane half 0 -> K 0..15 contiguous, half 1 -> +16.
__device__ __forceinline__ v16bf load_frag_a(const unsigned int* __restrict__ row, int ku) {
    v8u t;
#pragma unroll
    for (int j = 0; j < 8; ++j) {
        t[j]     = row[ku + j];
    }
    return __builtin_bit_cast(v16bf, t);
}
// B-matrix 32x16 bf16 fragment: lane half 0 -> K 0..15 contiguous, half 1 -> +16.
__device__ __forceinline__ v16bf load_frag_b(const unsigned int* __restrict__ row, int ku) {
    v8u t;
#pragma unroll
    for (int j = 0; j < 8; ++j) t[j] = row[ku + j];
    return __builtin_bit_cast(v16bf, t);
}

// ============================ pre-pass kernels =============================

// X[m,k] -> (x - mu), bf16 split, stored as interleaved pairs {hiPair, loPair}
__global__ __launch_bounds__(256)
void split_x_kernel(const float* __restrict__ x, const float* __restrict__ mu,
                    uint2* __restrict__ XHL)
{
    size_t pidx = ((size_t)blockIdx.x * 256 + threadIdx.x) * 2;   // pair index
#pragma unroll
    for (int j = 0; j < 2; ++j) {
        size_t pi = pidx + j;
        size_t i  = pi * 2;
        int k = (int)(i & (DIM_I - 1));
        float v0 = x[i]     - mu[k];
        float v1 = x[i + 1] - mu[k + 1];
        unsigned short h0 = f2bf(v0), h1 = f2bf(v1);
        unsigned short l0 = f2bf(v0 - bf2f(h0)), l1 = f2bf(v1 - bf2f(h1));
        XHL[pi] = make_uint2(pack2(h0, h1), pack2(l0, l1));
    }
}

// W[k,col] -> WT[col,k] split into bf16 hi/lo planes (32x32 LDS-tiled transpose)
// WTlo must be placed exactly DIM_O*DIM_I elements after WThi (one 3D tensor).
__global__ __launch_bounds__(256)
void split_wT_kernel(const float* __restrict__ W,
                     unsigned short* __restrict__ WThi,
                     unsigned short* __restrict__ WTlo)
{
    __shared__ unsigned short th[32][33];
    __shared__ unsigned short tl[32][33];
    const int col0 = blockIdx.x * 32;
    const int k0   = blockIdx.y * 32;
    const int tx = threadIdx.x;           // 0..31
    const int ty = threadIdx.y;           // 0..7
#pragma unroll
    for (int i = 0; i < 4; ++i) {
        const int k = k0 + ty + i * 8;
        float v = W[(size_t)k * DIM_O + col0 + tx];
        unsigned short h = f2bf(v);
        th[ty + i * 8][tx] = h;
        tl[ty + i * 8][tx] = f2bf(v - bf2f(h));
    }
    __syncthreads();
#pragma unroll
    for (int i = 0; i < 4; ++i) {
        const int col = col0 + ty + i * 8;
        WThi[(size_t)col * DIM_I + k0 + tx] = th[tx][ty + i * 8];
        WTlo[(size_t)col * DIM_I + k0 + tx] = tl[tx][ty + i * 8];
    }
}

// ============================ main GEMM kernel =============================

__global__ __launch_bounds__(256)
void cwic_wmma_ws_kernel(const uint2* __restrict__ XHL,
                         const unsigned short* __restrict__ WThi,
                         const unsigned short* __restrict__ WTlo,
                         const float* __restrict__ thr,
                         const float* __restrict__ out_mu,
                         float* __restrict__ y)
{
    // [buf][hi/lo plane][row][k-pair]; B planes contiguous for one TDM tile
    __shared__ unsigned int As[2][2][MT][LDP];
    __shared__ unsigned int Bs[2][2][NT][LDP];

    const int n_stripe = blockIdx.x;
    const int m_block  = blockIdx.y * MT;
    const int tid      = threadIdx.x;
    const int lane     = tid & 31;
    const int wave     = tid >> 5;
    const int wm       = wave & 3;
    const int wn       = wave >> 2;
    const int lm       = lane & 15;
    const int khalf    = lane >> 4;

    v8f acc[2][4];
#pragma unroll
    for (int i = 0; i < 2; ++i)
#pragma unroll
        for (int j = 0; j < 4; ++j)
#pragma unroll
            for (int e = 0; e < 8; ++e) acc[i][j][e] = 0.0f;

    // ---- B stage: copy pre-split WT tile (64k x 128q x 2 planes) into LDS
    auto stage_b = [&](int k0, int p) {
#if CWIC_USE_TDM
        if (wave == 0) {
            const unsigned long long gaddr =
                (unsigned long long)(uintptr_t)WThi +
                ((size_t)(n_stripe * QW) * DIM_I + (size_t)k0) * 2;
            const unsigned int lds = (unsigned int)(uintptr_t)&Bs[p][0][0][0];
            // D# group0: count=1 | lds_addr | global_addr(57b) | type=2
            v4u g0 = { 1u, lds, (unsigned int)gaddr,
                       (unsigned int)(gaddr >> 32) | (2u << 30) };
            // D# group1: data_size=2B, pad_enable, pad_interval=32dw, pad_amount=1dw
            //   tensor_dim0=4096, tensor_dim1=4096, tile_dim0=64, tile_dim1=128,
            //   tile_dim2=2, dim0_stride=4096, dim1_stride=4096*4096 (planes)
            v8i g1 = { 0x01110000, 0x10000000, 0x10000000, (64 << 16),
                       128 | (2 << 16), 4096, 0, 0x100 };
            v4i g2 = { 2, 0, 0, 0 };   // tensor_dim2 = 2 planes
            v4i g3 = { 0, 0, 0, 0 };
            __builtin_amdgcn_tensor_load_to_lds(g0, g1, g2, g3, 0);
        }
#else
        const int chunk = tid & 7;             // 16B chunk along k
        const int r0    = tid >> 3;            // 32 rows per pass
#pragma unroll
        for (int pass = 0; pass < 4; ++pass) {
            const int q   = pass * 32 + r0;
            const int col = n_stripe * QW + q;
            const size_t goff = ((size_t)col * DIM_I + (size_t)k0 + (size_t)chunk * 8) * 2;
            unsigned long long gH = (unsigned long long)(uintptr_t)WThi + goff;
            unsigned long long gL = (unsigned long long)(uintptr_t)WTlo + goff;
            unsigned int ldsH = (unsigned int)(uintptr_t)&Bs[p][0][q][chunk * 4];
            unsigned int ldsL = (unsigned int)(uintptr_t)&Bs[p][1][q][chunk * 4];
            asm volatile("global_load_async_to_lds_b128 %0, %1, off"
                         :: "v"(ldsH), "v"(gH) : "memory");
            asm volatile("global_load_async_to_lds_b128 %0, %1, off"
                         :: "v"(ldsL), "v"(gL) : "memory");
        }
#endif
    };

    auto wait_b = [&]() {
#if CWIC_USE_TDM
        if (wave == 0) __builtin_amdgcn_s_wait_tensorcnt((short)0);
#else
        asm volatile("s_wait_asynccnt 0" ::: "memory");
#endif
    };

    // ---- A stage: load interleaved pre-split pairs, gate vs thresholds[.,n]
    auto stage_a = [&](int k0, int p) {
        const int kp = tid & 31;
        const int r0 = tid >> 5;
        const int kk = k0 + 2 * kp;
        const float t0 = thr[(size_t)kk * NSTRIPE + n_stripe];
        const float t1 = thr[(size_t)(kk + 1) * NSTRIPE + n_stripe];
#pragma unroll
        for (int pp = 0; pp < MT / 8; ++pp) {
            const int m = pp * 8 + r0;
            const uint2 hl = XHL[(size_t)(m_block + m) * (DIM_I / 2) + (size_t)(k0 >> 1) + kp];
            const unsigned int hp = hl.x;
            const unsigned int lp = hl.y;
            const float x0 = bf2f((unsigned short)hp) + bf2f((unsigned short)lp);
            const float x1 = bf2f((unsigned short)(hp >> 16)) + bf2f((unsigned short)(lp >> 16));
            const unsigned int msk =
                (__builtin_fabsf(x0) >= t0 ? 0x0000FFFFu : 0u) |
                (__builtin_fabsf(x1) >= t1 ? 0xFFFF0000u : 0u);
            As[p][0][m][kp] = hp & msk;
            As[p][1][m][kp] = lp & msk;
        }
    };

    auto compute = [&](int p) {
#pragma unroll
        for (int ks = 0; ks < 2; ++ks) {
            const int kua = ks * 16 + khalf * 8;
            const int kub = ks * 16 + khalf * 8;
            v16bf aH[2], aL[2], bH[4], bL[4];
#pragma unroll
            for (int mt = 0; mt < 2; ++mt) {
                const int r = wm * 32 + mt * 16 + lm;
                aH[mt] = load_frag_a(As[p][0][r], kua);
                aL[mt] = load_frag_a(As[p][1][r], kua);
            }
#pragma unroll
            for (int nt = 0; nt < 4; ++nt) {
                const int r = wn * 64 + nt * 16 + lm;
                bH[nt] = load_frag_b(Bs[p][0][r], kub);
                bL[nt] = load_frag_b(Bs[p][1][r], kub);
            }
#pragma unroll
            for (int mt = 0; mt < 2; ++mt)
#pragma unroll
                for (int nt = 0; nt < 4; ++nt) {
                    acc[mt][nt] = __builtin_amdgcn_wmma_f32_16x16x32_bf16(
                        false, aH[mt], false, bH[nt], (short)0, acc[mt][nt], false, false);
                    acc[mt][nt] = __builtin_amdgcn_wmma_f32_16x16x32_bf16(
                        false, aH[mt], false, bL[nt], (short)0, acc[mt][nt], false, false);
                    acc[mt][nt] = __builtin_amdgcn_wmma_f32_16x16x32_bf16(
                        false, aL[mt], false, bH[nt], (short)0, acc[mt][nt], false, false);
                }
        }
    };

    // ---- software pipeline: double-buffered LDS, 1 barrier / K-tile
    stage_b(0, 0);
    stage_a(0, 0);
    wait_b();
    __syncthreads();

    const int NTILES = DIM_I / KT;
    for (int t = 0; t < NTILES; ++t) {
        const int p = t & 1;
        if (t + 1 < NTILES) {
            stage_b((t + 1) * KT, p ^ 1);
            stage_a((t + 1) * KT, p ^ 1);
        }
        compute(p);
        wait_b();
        __syncthreads();
    }

    // ---- epilogue: C/D layout -> M = vgpr + 8*lanehalf, N = lane&15
    const int colbase = n_stripe * QW + wn * 64;
#pragma unroll
    for (int nt = 0; nt < 4; ++nt) {
        const int col = colbase + nt * 16 + lm;
        const float om = out_mu[col];
#pragma unroll
        for (int mt = 0; mt < 2; ++mt) {
            const int mrow = m_block + wm * 32 + mt * 16 + khalf * 8;
#pragma unroll
            for (int e = 0; e < 8; ++e) {
                y[(size_t)(mrow + e) * DIM_O + col] = acc[mt][nt][e] + om;
            }
        }
    }
}

// ==================== fallback (self-contained, round-1) ====================

__global__ __launch_bounds__(256)
void cwic_wmma_fallback(const float* __restrict__ x,
                        const float* __restrict__ W,
                        const float* __restrict__ thr,
                        const float* __restrict__ mu,
                        const float* __restrict__ out_mu,
                        float* __restrict__ y)
{
    __shared__ unsigned int As_hi[MT][LDP];
    __shared__ unsigned int As_lo[MT][LDP];
    __shared__ unsigned int Bs_hi[NT][LDP];
    __shared__ unsigned int Bs_lo[NT][LDP];

    const int n_stripe = blockIdx.x;
    const int m_block  = blockIdx.y * MT;
    const int tid      = threadIdx.x;
    const int lane     = tid & 31;
    const int wave     = tid >> 5;
    const int wm       = wave & 3;
    const int wn       = wave >> 2;
    const int lm       = lane & 15;
    const int khalf    = lane >> 4;

    const float* __restrict__ xrow = x + (size_t)m_block * DIM_I;
    const float* __restrict__ Wcol = W + (size_t)n_stripe * QW;

    v8f acc[2][4];
#pragma unroll
    for (int i = 0; i < 2; ++i)
#pragma unroll
        for (int j = 0; j < 4; ++j)
#pragma unroll
            for (int e = 0; e < 8; ++e) acc[i][j][e] = 0.0f;

    for (int k0 = 0; k0 < DIM_I; k0 += KT) {
        {
            const int kp = tid & 31;
            const int r0 = tid >> 5;
            const int kk = k0 + 2 * kp;
            const float t0 = thr[(size_t)kk * NSTRIPE + n_stripe];
            const float t1 = thr[(size_t)(kk + 1) * NSTRIPE + n_stripe];
            const float m0 = mu[kk];
            const float m1 = mu[kk + 1];
#pragma unroll
            for (int p = 0; p < MT / 8; ++p) {
                const int m = p * 8 + r0;
                float x0 = xrow[(size_t)m * DIM_I + kk]     - m0;
                float x1 = xrow[(size_t)m * DIM_I + kk + 1] - m1;
                float a0 = (__builtin_fabsf(x0) >= t0) ? x0 : 0.0f;
                float a1 = (__builtin_fabsf(x1) >= t1) ? x1 : 0.0f;
                unsigned short h0 = f2bf(a0), h1 = f2bf(a1);
                As_hi[m][kp] = pack2(h0, h1);
                As_lo[m][kp] = pack2(f2bf(a0 - bf2f(h0)), f2bf(a1 - bf2f(h1)));
            }
        }
        {
            const int q   = tid & 127;
            const int kp0 = tid >> 7;
#pragma unroll
            for (int p = 0; p < KT / 4; ++p) {
                const int kp = p * 2 + kp0;
                const int kk = k0 + 2 * kp;
                float w0 = Wcol[(size_t)kk * DIM_O + q];
                float w1 = Wcol[(size_t)(kk + 1) * DIM_O + q];
                unsigned short h0 = f2bf(w0), h1 = f2bf(w1);
                Bs_hi[q][kp] = pack2(h0, h1);
                Bs_lo[q][kp] = pack2(f2bf(w0 - bf2f(h0)), f2bf(w1 - bf2f(h1)));
            }
        }
        __syncthreads();

#pragma unroll
        for (int ks = 0; ks < 2; ++ks) {
            const int kua = ks * 16 + khalf * 8;
            const int kub = ks * 16 + khalf * 8;
            v16bf aH[2], aL[2], bH[4], bL[4];
#pragma unroll
            for (int mt = 0; mt < 2; ++mt) {
                const int r = wm * 32 + mt * 16 + lm;
                aH[mt] = load_frag_a(As_hi[r], kua);
                aL[mt] = load_frag_a(As_lo[r], kua);
            }
#pragma unroll
            for (int nt = 0; nt < 4; ++nt) {
                const int r = wn * 64 + nt * 16 + lm;
                bH[nt] = load_frag_b(Bs_hi[r], kub);
                bL[nt] = load_frag_b(Bs_lo[r], kub);
            }
#pragma unroll
            for (int mt = 0; mt < 2; ++mt)
#pragma unroll
                for (int nt = 0; nt < 4; ++nt) {
                    acc[mt][nt] = __builtin_amdgcn_wmma_f32_16x16x32_bf16(
                        false, aH[mt], false, bH[nt], (short)0, acc[mt][nt], false, false);
                    acc[mt][nt] = __builtin_amdgcn_wmma_f32_16x16x32_bf16(
                        false, aH[mt], false, bL[nt], (short)0, acc[mt][nt], false, false);
                    acc[mt][nt] = __builtin_amdgcn_wmma_f32_16x16x32_bf16(
                        false, aL[mt], false, bH[nt], (short)0, acc[mt][nt], false, false);
                }
        }
        __syncthreads();
    }

    const int colbase = n_stripe * QW + wn * 64;
#pragma unroll
    for (int nt = 0; nt < 4; ++nt) {
        const int col = colbase + nt * 16 + lm;
        const float om = out_mu[col];
#pragma unroll
        for (int mt = 0; mt < 2; ++mt) {
            const int mrow = m_block + wm * 32 + mt * 16 + khalf * 8;
#pragma unroll
            for (int e = 0; e < 8; ++e) {
                y[(size_t)(mrow + e) * DIM_O + col] = acc[mt][nt][e] + om;
            }
        }
    }
}

// ================================ launcher =================================

extern "C" void kernel_launch(void* const* d_in, const int* in_sizes, int n_in,
                              void* d_out, int out_size, void* d_ws, size_t ws_size,
                              hipStream_t stream) {
    (void)in_sizes; (void)n_in; (void)out_size;
    const float* x      = (const float*)d_in[0];
    const float* W      = (const float*)d_in[1];
    const float* thr    = (const float*)d_in[2];
    const float* mu     = (const float*)d_in[3];
    const float* out_mu = (const float*)d_in[4];
    float* y = (float*)d_out;

    const size_t XSZ = (size_t)DIM_M * DIM_I * sizeof(unsigned short);  // 8 MB (x2 interleaved)
    const size_t WSZ = (size_t)DIM_O * DIM_I * sizeof(unsigned short);  // 32 MB per plane
    const size_t need = 2 * XSZ + 2 * WSZ;                              // 80 MB

    if (ws_size >= need && d_ws != nullptr) {
        uint2*          XHL  = (uint2*)d_ws;                            // interleaved hi/lo pairs
        unsigned short* WThi = (unsigned short*)((char*)d_ws + 2 * XSZ);
        unsigned short* WTlo = WThi + (size_t)DIM_O * DIM_I;            // contiguous plane

        split_x_kernel<<<(DIM_M * DIM_I / 2) / (256 * 2), 256, 0, stream>>>(x, mu, XHL);
        split_wT_kernel<<<dim3(DIM_O / 32, DIM_I / 32), dim3(32, 8), 0, stream>>>(W, WThi, WTlo);
        cwic_wmma_ws_kernel<<<dim3(NSTRIPE, DIM_M / MT), 256, 0, stream>>>(
            XHL, WThi, WTlo, thr, out_mu, y);
    } else {
        cwic_wmma_fallback<<<dim3(NSTRIPE, DIM_M / MT), 256, 0, stream>>>(
            x, W, thr, mu, out_mu, y);
    }
}
